// DefectDetection_19421842112784
// MI455X (gfx1250) — compile-verified, matrix operand order checked by reading, not run on hardware
//
#include <hip/hip_runtime.h>
#include <hip/hip_fp16.h>

// ---------------------------------------------------------------------------
// DefectDetection GNN forward for MI455X (gfx1250, wave32, WMMA).
// Dims from reference: N=512, E=8192, HID=64, NH=16, OUT=128, D1=2048,
// JUMP=2240, LH=128, NCLS=2.  Workspace use ~40MB.
// All WMMA B-operands are kept N-major (transposed) so every fragment load is
// a contiguous 32B v16h (2x global_load_b128 per operand) instead of 16 b16
// gathers + v_mov_b16 merges.
// ---------------------------------------------------------------------------

typedef __attribute__((ext_vector_type(16))) _Float16 v16h;
typedef __attribute__((ext_vector_type(8)))  _Float16 v8h;
typedef __attribute__((ext_vector_type(8)))  float    v8f;

#define KN    512
#define KE    8192
#define KHID  64
#define KNH   16
#define KOUT  128
#define KD1   2048
#define KN2   256
#define KN4   128
#define KJUMP 2240
#define KLH   128

__device__ __forceinline__ float sigf(float x)  { return 1.0f / (1.0f + expf(-x)); }
__device__ __forceinline__ float leakyf(float x){ return x > 0.0f ? x : 0.2f * x; }
__device__ __forceinline__ float eluf(float x)  { return x > 0.0f ? x : expf(x) - 1.0f; }

// One wave computes a 16x16 f32 tile of A(MxK,row-major f16) * B(KxN) where B
// is supplied N-major (BT[n][k]).  Contiguous k order, same in A and B (wave32): lane<16
// holds K=[0,16), lane>=16 holds K=[16,32); lane&15 selects A-row / B-col.
__device__ __forceinline__ v8f wmma_tile(const _Float16* __restrict__ A, int lda,
                                         const _Float16* __restrict__ BT, int ldbt,
                                         int K, int lane) {
  v8f c = {};
  const int rc   = lane & 15;
  const int half = lane >> 4;
  const _Float16* ap = A + (size_t)rc * lda + half * 16;
  const _Float16* bp = BT + (size_t)rc * ldbt + half * 16;
  for (int k0 = 0; k0 < K; k0 += 32) {
    v16h a = *(const v16h*)(ap + k0);
    v16h b = *(const v16h*)(bp + k0);
    __builtin_prefetch(ap + k0 + 32, 0, 3);   // global_prefetch_b8
    __builtin_prefetch(bp + k0 + 32, 0, 3);
    c = __builtin_amdgcn_wmma_f32_16x16x32_f16(false, a, false, b, (short)0, c,
                                               false, false);
  }
  return c;
}

// ----------------------------- utility kernels ------------------------------

__global__ void fzero_kernel(float* p, int n) {
  int i = blockIdx.x * blockDim.x + threadIdx.x;
  if (i < n) p[i] = 0.0f;
}
__global__ void ifill_kernel(int* p, int n, int v) {
  int i = blockIdx.x * blockDim.x + threadIdx.x;
  if (i < n) p[i] = v;
}
__global__ void f2h_kernel(const float* __restrict__ s, _Float16* __restrict__ d, int n) {
  int i = blockIdx.x * blockDim.x + threadIdx.x;
  if (i < n) d[i] = (_Float16)s[i];
}
// batched f32[R][C] -> f16 transposed [C][R]
__global__ void f2ht_kernel(const float* __restrict__ s, _Float16* __restrict__ d,
                            int R, int C, int total) {
  int idx = blockIdx.x * blockDim.x + threadIdx.x;
  if (idx >= total) return;
  int b = idx / (R * C), rem = idx % (R * C);
  int r = rem / C, c = rem % C;
  d[(size_t)b * R * C + (size_t)c * R + r] = (_Float16)s[idx];
}

// --------------------- stage 0: single-node attention -----------------------
// h = elu(sigmoid(leaky(Wh0 . a_sn)) * Wh0),  Wh0 = features @ W_sn
__global__ void sna_kernel(const float* __restrict__ feat, const float* __restrict__ Wsn,
                           const float* __restrict__ asn, float* __restrict__ h64,
                           _Float16* __restrict__ h64h) {
  __shared__ float wh0[KHID];
  __shared__ float red[KHID];
  int nidx = blockIdx.x, j = threadIdx.x;  // 64 threads
  float acc = 0.0f;
  for (int d = 0; d < KHID; ++d) acc += feat[nidx * KHID + d] * Wsn[d * KHID + j];
  wh0[j] = acc;
  red[j] = acc * asn[j];
  __syncthreads();
  for (int s = 32; s > 0; s >>= 1) {
    if (j < s) red[j] += red[j + s];
    __syncthreads();
  }
  float gate = sigf(leakyf(red[0]));
  float v = eluf(gate * wh0[j]);
  h64[nidx * KHID + j] = v;
  h64h[nidx * KHID + j] = (_Float16)v;
}

// ------------------------- global attention pool ----------------------------
// out[D] = sum_i softmax_i(sigmoid(x_i.Wg+bg)) * x_i   (single block, 512 thr)
__global__ void gpool_kernel(const float* __restrict__ x, int n, int D,
                             const float* __restrict__ Wg, const float* __restrict__ bg,
                             float* __restrict__ out) {
  __shared__ float sg[512];
  __shared__ float red[512];
  int tid = threadIdx.x;
  float g = -1e30f;
  if (tid < n) {
    float acc = 0.0f;
    const float* xr = x + (size_t)tid * D;
    for (int d = 0; d < D; ++d) acc += xr[d] * Wg[d];
    g = sigf(acc + bg[0]);
  }
  red[tid] = g;
  __syncthreads();
  for (int s = 256; s > 0; s >>= 1) {
    if (tid < s) red[tid] = fmaxf(red[tid], red[tid + s]);
    __syncthreads();
  }
  float mx = red[0];
  __syncthreads();
  float e = (tid < n) ? expf(g - mx) : 0.0f;
  red[tid] = e;
  __syncthreads();
  for (int s = 256; s > 0; s >>= 1) {
    if (tid < s) red[tid] += red[tid + s];
    __syncthreads();
  }
  float inv = 1.0f / red[0];
  __syncthreads();
  sg[tid] = e * inv;
  __syncthreads();
  for (int d = tid; d < D; d += blockDim.x) {
    float acc = 0.0f;
    for (int i = 0; i < n; ++i) acc += sg[i] * x[(size_t)i * D + d];
    out[d] = acc;
  }
}

// --------------------- WMMA GEMM: Wh = h @ W_gat (per head) -----------------
// A: h64h [512x64]; BT: WghT [h][128][64]; out f16 transposed WhT [h][128][512]
__global__ void wh_gemm_kernel(const _Float16* __restrict__ A,
                               const _Float16* __restrict__ WghT,
                               _Float16* __restrict__ WhT) {
  int wid = threadIdx.x >> 5, lane = threadIdx.x & 31;
  int t = blockIdx.x * 8 + wid;  // 4096 tiles
  int mt = t >> 7, rem = t & 127;
  int h = rem >> 3, ot = rem & 7;
  v8f c = wmma_tile(A + mt * 16 * KHID, KHID,
                    WghT + ((size_t)h * KOUT + ot * 16) * KHID, KHID, KHID, lane);
  int half = lane >> 4, col = lane & 15;
  v8h pk;
#pragma unroll
  for (int r = 0; r < 8; ++r) pk[r] = (_Float16)c[r];
  // transposed store: 8 contiguous f16 (rows mt*16+8*half..+7 of column ot*16+col)
  *(v8h*)(WhT + ((size_t)h * KOUT + ot * 16 + col) * KN + mt * 16 + 8 * half) = pk;
}

// s1[h,n] = Wh[h,n,:].a1[h]; s2 likewise (reads transposed WhT, coalesced in n)
__global__ void svec_kernel(const _Float16* __restrict__ WhT, const float* __restrict__ a1,
                            const float* __restrict__ a2, float* __restrict__ s1,
                            float* __restrict__ s2) {
  int idx = blockIdx.x * blockDim.x + threadIdx.x;  // 8192
  int h = idx >> 9, n = idx & (KN - 1);
  const _Float16* w = WhT + (size_t)h * KOUT * KN + n;
  float x1 = 0.0f, x2 = 0.0f;
  for (int o = 0; o < KOUT; ++o) {
    float v = (float)w[(size_t)o * KN];
    x1 += v * a1[h * KOUT + o];
    x2 += v * a2[h * KOUT + o];
  }
  s1[idx] = x1;
  s2[idx] = x2;
}

// ---- deterministic scatter (emulates XLA .set last-writer semantics) -------
__global__ void win_kernel(const int* __restrict__ ei, int E, int shift, int n,
                           int* __restrict__ win) {
  int e = blockIdx.x * blockDim.x + threadIdx.x;
  if (e >= E) return;
  int s = ei[e] >> shift, d = ei[E + e] >> shift;
  atomicMax(&win[s * n + d], e);
}

// e3[h,s,d] = edgesAttr[winner].a3_gat[h]
__global__ void e3_fill_kernel(const int* __restrict__ ei, const int* __restrict__ win,
                               const float* __restrict__ eAttr, const float* __restrict__ a3g,
                               float* __restrict__ e3) {
  int idx = blockIdx.x * blockDim.x + threadIdx.x;  // 16*8192
  int h = idx >> 13, e = idx & (KE - 1);
  int cell = ei[e] * KN + ei[KE + e];
  if (win[cell] != e) return;
  float z = 0.0f;
  for (int k = 0; k < KHID; ++k) z += eAttr[e * KHID + k] * a3g[h * KHID + k];
  e3[(size_t)h * KN * KN + cell] = z;
}

// ------------------ attention rows: softmax over neighbors ------------------
__global__ void att_kernel(const float* __restrict__ s1, const float* __restrict__ s2,
                           const float* __restrict__ e3, const float* __restrict__ adj,
                           _Float16* __restrict__ att) {
  int i = blockIdx.x, h = blockIdx.y, tid = threadIdx.x;  // 256 thr
  __shared__ float row[KN];
  __shared__ float red[256];
  const float* e3r = e3 + ((size_t)h * KN + i) * KN;
  const float* ar = adj + (size_t)i * KN;
  float s1v = s1[h * KN + i];
  for (int j = tid; j < KN; j += 256) {
    float ev = leakyf(s1v + s2[h * KN + j] + e3r[j]);
    row[j] = (ar[j] > 0.0f) ? ev : -1e9f;
  }
  __syncthreads();
  red[tid] = fmaxf(row[tid], row[tid + 256]);
  __syncthreads();
  for (int s = 128; s > 0; s >>= 1) {
    if (tid < s) red[tid] = fmaxf(red[tid], red[tid + s]);
    __syncthreads();
  }
  float mx = red[0];
  __syncthreads();
  float e0 = expf(row[tid] - mx), e1 = expf(row[tid + 256] - mx);
  red[tid] = e0 + e1;
  __syncthreads();
  for (int s = 128; s > 0; s >>= 1) {
    if (tid < s) red[tid] += red[tid + s];
    __syncthreads();
  }
  float inv = 1.0f / red[0];
  _Float16* ao = att + ((size_t)h * KN + i) * KN;
  ao[tid] = (_Float16)(e0 * inv);
  ao[tid + 256] = (_Float16)(e1 * inv);
}

// ---------------- WMMA GEMM: h2048 = elu(att @ Wh) per head -----------------
// A: att[h] [512x512]; BT: WhT[h] [128][512]; K=512
__global__ void av_gemm_kernel(const _Float16* __restrict__ att,
                               const _Float16* __restrict__ WhT,
                               float* __restrict__ h2048) {
  int wid = threadIdx.x >> 5, lane = threadIdx.x & 31;
  int t = blockIdx.x * 8 + wid;  // 4096 tiles
  int mt = t >> 7, rem = t & 127;
  int h = rem >> 3, ot = rem & 7;
  v8f c = wmma_tile(att + (size_t)h * KN * KN + mt * 16 * KN, KN,
                    WhT + ((size_t)h * KOUT + ot * 16) * KN, KN, KN, lane);
  int half = lane >> 4, col = lane & 15;
#pragma unroll
  for (int r = 0; r < 8; ++r) {
    int rowo = mt * 16 + r + 8 * half;
    h2048[(size_t)rowo * KD1 + h * KOUT + ot * 16 + col] = eluf(c[r]);
  }
}

// -- WMMA fused: ea3[e] = elu(edgesAttr @ We_gat).flatten_per_row . a3_o -----
__global__ void ea3_kernel(const _Float16* __restrict__ eAh,
                           const _Float16* __restrict__ WegT,
                           const float* __restrict__ a3o, float* __restrict__ ea3) {
  int wid = threadIdx.x >> 5, lane = threadIdx.x & 31;
  int mt = blockIdx.x * 8 + wid;  // 512 m-tiles of edges
  int half = lane >> 4, col = lane & 15;
  float acc[8];
#pragma unroll
  for (int r = 0; r < 8; ++r) acc[r] = 0.0f;
  const _Float16* Ap = eAh + (size_t)mt * 16 * KHID;
  for (int nt = 0; nt < 128; ++nt) {
    int h = nt >> 3, ol = (nt & 7) * 16;
    v8f c = wmma_tile(Ap, KHID, WegT + ((size_t)h * KOUT + ol) * KHID, KHID, KHID, lane);
    float av = a3o[h * KOUT + ol + col];
#pragma unroll
    for (int r = 0; r < 8; ++r) acc[r] += eluf(c[r]) * av;
  }
#pragma unroll
  for (int r = 0; r < 8; ++r)
    for (int m = 1; m < 16; m <<= 1) acc[r] += __shfl_xor(acc[r], m, 32);
  if (col == 0) {
#pragma unroll
    for (int r = 0; r < 8; ++r) ea3[mt * 16 + r + 8 * half] = acc[r];
  }
}

// -------------------------- edge pooling 1 ----------------------------------
__global__ void epool1_kernel(const float* __restrict__ h2048, const float* __restrict__ Wp1,
                              const float* __restrict__ bp1, float* __restrict__ hp,
                              _Float16* __restrict__ hph) {
  __shared__ float red[256];
  int m = blockIdx.x, tid = threadIdx.x;  // 256 thr
  const float* xr = h2048 + (size_t)m * 2 * KD1;
  float acc = 0.0f;
  for (int k = tid; k < 2 * KD1; k += 256) acc += xr[k] * Wp1[k];
  red[tid] = acc;
  __syncthreads();
  for (int s = 128; s > 0; s >>= 1) {
    if (tid < s) red[tid] += red[tid + s];
    __syncthreads();
  }
  float sc = sigf(red[0] + bp1[0]);
  for (int d = tid; d < KD1; d += 256) {
    float v = (xr[d] + xr[KD1 + d]) * sc;
    hp[(size_t)m * KD1 + d] = v;
    hph[(size_t)m * KD1 + d] = (_Float16)v;
  }
}

// ----------- pooled graph: adj2 (.add) + e3b (.set, deterministic) ----------
__global__ void pool2_scatter_kernel(const int* __restrict__ ei, const int* __restrict__ win2,
                                     const float* __restrict__ ea3, float* __restrict__ adj2,
                                     float* __restrict__ e3b) {
  int e = blockIdx.x * blockDim.x + threadIdx.x;
  if (e >= KE) return;
  int cell = (ei[e] >> 1) * KN2 + (ei[KE + e] >> 1);
  atomicAdd(&adj2[cell], 1.0f);
  if (win2[cell] == e) e3b[cell] = ea3[e];
}

// ---------- WMMA GEMM: Wh2 = hp @ Wo  [256x2048]x[2048x128], BT = WoT -------
__global__ void wh2_gemm_kernel(const _Float16* __restrict__ hph,
                                const _Float16* __restrict__ WoT,
                                float* __restrict__ Wh2) {
  int wid = threadIdx.x >> 5, lane = threadIdx.x & 31;
  int t = blockIdx.x * 8 + wid;  // 128 tiles
  int mt = t >> 3, ot = t & 7;
  v8f c = wmma_tile(hph + (size_t)mt * 16 * KD1, KD1,
                    WoT + (size_t)(ot * 16) * KD1, KD1, KD1, lane);
  int half = lane >> 4, col = lane & 15;
#pragma unroll
  for (int r = 0; r < 8; ++r)
    Wh2[(size_t)(mt * 16 + r + 8 * half) * KOUT + ot * 16 + col] = c[r];
}

__global__ void svec2_kernel(const float* __restrict__ Wh2, const float* __restrict__ a1o,
                             const float* __restrict__ a2o, float* __restrict__ s1o,
                             float* __restrict__ s2o) {
  int n = threadIdx.x;  // 256
  float x1 = 0.0f, x2 = 0.0f;
  for (int o = 0; o < KOUT; ++o) {
    float v = Wh2[n * KOUT + o];
    x1 += v * a1o[o];
    x2 += v * a2o[o];
  }
  s1o[n] = x1;
  s2o[n] = x2;
}

// ------------- second attention (1 head, no elu) + matvec -------------------
__global__ void att2_kernel(const float* __restrict__ s1o, const float* __restrict__ s2o,
                            const float* __restrict__ e3b, const float* __restrict__ adj2,
                            const float* __restrict__ Wh2, float* __restrict__ h3) {
  __shared__ float row[KN2];
  __shared__ float red[KN2];
  int i = blockIdx.x, tid = threadIdx.x;  // 256 thr
  float ev = leakyf(s1o[i] + s2o[tid] + e3b[i * KN2 + tid]);
  row[tid] = (adj2[i * KN2 + tid] > 0.0f) ? ev : -1e9f;
  __syncthreads();
  red[tid] = row[tid];
  __syncthreads();
  for (int s = 128; s > 0; s >>= 1) {
    if (tid < s) red[tid] = fmaxf(red[tid], red[tid + s]);
    __syncthreads();
  }
  float mx = red[0];
  __syncthreads();
  float e = expf(row[tid] - mx);
  red[tid] = e;
  __syncthreads();
  for (int s = 128; s > 0; s >>= 1) {
    if (tid < s) red[tid] += red[tid + s];
    __syncthreads();
  }
  float inv = 1.0f / red[0];
  __syncthreads();
  row[tid] = e * inv;
  __syncthreads();
  if (tid < KOUT) {
    float acc = 0.0f;
    for (int j = 0; j < KN2; ++j) acc += row[j] * Wh2[j * KOUT + tid];
    h3[i * KOUT + tid] = acc;
  }
}

// -------------------------- edge pooling 2 ----------------------------------
__global__ void epool2_kernel(const float* __restrict__ h3, const float* __restrict__ Wp2,
                              const float* __restrict__ bp2, float* __restrict__ h4) {
  __shared__ float red[128];
  int m = blockIdx.x, tid = threadIdx.x;  // 128 thr
  const float* xr = h3 + (size_t)m * 2 * KOUT;
  float acc = xr[tid] * Wp2[tid] + xr[tid + 128] * Wp2[tid + 128];
  red[tid] = acc;
  __syncthreads();
  for (int s = 64; s > 0; s >>= 1) {
    if (tid < s) red[tid] += red[tid + s];
    __syncthreads();
  }
  float sc = sigf(red[0] + bp2[0]);
  h4[m * KOUT + tid] = (xr[tid] + xr[KOUT + tid]) * sc;
}

// --------- head: 2-layer bi-LSTM (T=1, h0=c0=0 -> Whh unused) + fc ----------
__global__ void head_kernel(const float* __restrict__ hs, const float* __restrict__ Wih0,
                            const float* __restrict__ b0, const float* __restrict__ Wih1,
                            const float* __restrict__ b1, const float* __restrict__ fcW,
                            const float* __restrict__ fcb, float* __restrict__ out) {
  __shared__ float sx[KJUMP];
  __shared__ float gates[4 * KLH];
  __shared__ float y1[2 * KLH];
  __shared__ float y2[2 * KLH];
  int tid = threadIdx.x;  // 256 thr
  for (int k = tid; k < KJUMP; k += 256) sx[k] = hs[k];
  __syncthreads();
  for (int dir = 0; dir < 2; ++dir) {
    for (int g = tid; g < 4 * KLH; g += 256) {
      float acc = b0[dir * 4 * KLH + g];
      const float* w = Wih0 + ((size_t)dir * 4 * KLH + g) * KJUMP;
      for (int k = 0; k < KJUMP; ++k) acc += w[k] * sx[k];
      gates[g] = acc;
    }
    __syncthreads();
    if (tid < KLH) {
      float ci = sigf(gates[tid]) * tanhf(gates[2 * KLH + tid]);
      y1[dir * KLH + tid] = sigf(gates[3 * KLH + tid]) * tanhf(ci);
    }
    __syncthreads();
  }
  for (int dir = 0; dir < 2; ++dir) {
    for (int g = tid; g < 4 * KLH; g += 256) {
      float acc = b1[dir * 4 * KLH + g];
      const float* w = Wih1 + ((size_t)dir * 4 * KLH + g) * (2 * KLH);
      for (int k = 0; k < 2 * KLH; ++k) acc += w[k] * y1[k];
      gates[g] = acc;
    }
    __syncthreads();
    if (tid < KLH) {
      float ci = sigf(gates[tid]) * tanhf(gates[2 * KLH + tid]);
      y2[dir * KLH + tid] = sigf(gates[3 * KLH + tid]) * tanhf(ci);
    }
    __syncthreads();
  }
  if (tid == 0) {
    float l0 = fcb[0], l1 = fcb[1];
    for (int k = 0; k < 2 * KLH; ++k) {
      l0 += y2[k] * fcW[k * 2 + 0];
      l1 += y2[k] * fcW[k * 2 + 1];
    }
    float m = fmaxf(l0, l1);
    float e0 = expf(l0 - m), e1 = expf(l1 - m);
    float inv = 1.0f / (e0 + e1);
    out[0] = e0 * inv;
    out[1] = e1 * inv;
  }
}

// ---------------------------------------------------------------------------

extern "C" void kernel_launch(void* const* d_in, const int* in_sizes, int n_in,
                              void* d_out, int out_size, void* d_ws, size_t ws_size,
                              hipStream_t stream) {
  const float* features = (const float*)d_in[0];
  const int*   edge_index = (const int*)d_in[1];
  const float* edgesAttr = (const float*)d_in[2];
  const float* adjacency = (const float*)d_in[3];
  // d_in[4] = node2node_features: provably redundant (scatter of edgesAttr) -> skipped.
  const float* W_sn = (const float*)d_in[5];
  const float* a_sn = (const float*)d_in[6];
  const float* W_gat = (const float*)d_in[7];
  const float* a1_gat = (const float*)d_in[8];
  const float* a2_gat = (const float*)d_in[9];
  const float* a3_gat = (const float*)d_in[10];
  const float* We_gat = (const float*)d_in[11];
  const float* Wg1 = (const float*)d_in[12];
  const float* bg1 = (const float*)d_in[13];
  const float* Wp1 = (const float*)d_in[14];
  const float* bp1 = (const float*)d_in[15];
  const float* Wg2 = (const float*)d_in[16];
  const float* bg2 = (const float*)d_in[17];
  const float* Wo = (const float*)d_in[18];
  const float* a1_o = (const float*)d_in[19];
  const float* a2_o = (const float*)d_in[20];
  const float* a3_o = (const float*)d_in[21];
  // d_in[22] = We_o: dead code in reference -> skipped.
  const float* Wp2 = (const float*)d_in[23];
  const float* bp2 = (const float*)d_in[24];
  const float* Wg3 = (const float*)d_in[25];
  const float* bg3 = (const float*)d_in[26];
  const float* Wih0 = (const float*)d_in[27];
  // d_in[28] = Whh0: unused (T=1, h0=0).
  const float* b0 = (const float*)d_in[29];
  const float* Wih1 = (const float*)d_in[30];
  // d_in[31] = Whh1: unused.
  const float* b1 = (const float*)d_in[32];
  const float* fc_W = (const float*)d_in[33];
  const float* fc_b = (const float*)d_in[34];
  float* out = (float*)d_out;

  size_t off = 0;
  auto alloc = [&](size_t bytes) -> void* {
    void* p = (char*)d_ws + off;
    off += (bytes + 255) & ~(size_t)255;
    return p;
  };
  float*    h64   = (float*)alloc(KN * KHID * 4);
  _Float16* h64h  = (_Float16*)alloc(KN * KHID * 2);
  _Float16* WghT  = (_Float16*)alloc(KNH * KHID * KOUT * 2);   // [h][128][64]
  _Float16* WegT  = (_Float16*)alloc(KNH * KHID * KOUT * 2);   // [h][128][64]
  _Float16* eAh   = (_Float16*)alloc(KE * KHID * 2);
  _Float16* WoT   = (_Float16*)alloc(KD1 * KOUT * 2);          // [128][2048]
  _Float16* WhT   = (_Float16*)alloc((size_t)KNH * KN * KOUT * 2);  // [h][128][512]
  float*    s1    = (float*)alloc(KNH * KN * 4);
  float*    s2    = (float*)alloc(KNH * KN * 4);
  float*    e3    = (float*)alloc((size_t)KNH * KN * KN * 4);
  _Float16* att   = (_Float16*)alloc((size_t)KNH * KN * KN * 2);
  float*    h2048 = (float*)alloc((size_t)KN * KD1 * 4);
  float*    ea3   = (float*)alloc(KE * 4);
  float*    hp    = (float*)alloc((size_t)KN2 * KD1 * 4);
  _Float16* hph   = (_Float16*)alloc((size_t)KN2 * KD1 * 2);
  float*    adj2  = (float*)alloc(KN2 * KN2 * 4);
  float*    e3b   = (float*)alloc(KN2 * KN2 * 4);
  float*    Wh2   = (float*)alloc(KN2 * KOUT * 4);
  float*    s1o   = (float*)alloc(KN2 * 4);
  float*    s2o   = (float*)alloc(KN2 * 4);
  float*    h3    = (float*)alloc(KN2 * KOUT * 4);
  float*    h4    = (float*)alloc(KN4 * KOUT * 4);
  float*    hsbuf = (float*)alloc(KJUMP * 4);
  int*      win1  = (int*)alloc((size_t)KN * KN * 4);
  int*      win2  = (int*)alloc(KN2 * KN2 * 4);
  (void)ws_size; (void)in_sizes; (void)n_in; (void)out_size;

  // weight packs: f16, B-operands transposed to N-major
  f2ht_kernel<<<(KNH * KHID * KOUT + 255) / 256, 256, 0, stream>>>(W_gat, WghT, KHID, KOUT,
                                                                   KNH * KHID * KOUT);
  f2ht_kernel<<<(KNH * KHID * KOUT + 255) / 256, 256, 0, stream>>>(We_gat, WegT, KHID, KOUT,
                                                                   KNH * KHID * KOUT);
  f2ht_kernel<<<(KD1 * KOUT + 255) / 256, 256, 0, stream>>>(Wo, WoT, KD1, KOUT, KD1 * KOUT);
  f2h_kernel<<<(KE * KHID + 255) / 256, 256, 0, stream>>>(edgesAttr, eAh, KE * KHID);

  // scratch init (workspace is poisoned between runs)
  fzero_kernel<<<(KNH * KN * KN + 255) / 256, 256, 0, stream>>>(e3, KNH * KN * KN);
  fzero_kernel<<<(KN2 * KN2 + 255) / 256, 256, 0, stream>>>(adj2, KN2 * KN2);
  fzero_kernel<<<(KN2 * KN2 + 255) / 256, 256, 0, stream>>>(e3b, KN2 * KN2);
  ifill_kernel<<<(KN * KN + 255) / 256, 256, 0, stream>>>(win1, KN * KN, -1);
  ifill_kernel<<<(KN2 * KN2 + 255) / 256, 256, 0, stream>>>(win2, KN2 * KN2, -1);

  // stage 0: single-node attention
  sna_kernel<<<KN, KHID, 0, stream>>>(features, W_sn, a_sn, h64, h64h);
  gpool_kernel<<<1, 512, 0, stream>>>(h64, KN, KHID, Wg1, bg1, hsbuf);

  // multi-head GAT
  wh_gemm_kernel<<<512, 256, 0, stream>>>(h64h, WghT, WhT);          // WMMA
  svec_kernel<<<KNH * KN / 256, 256, 0, stream>>>(WhT, a1_gat, a2_gat, s1, s2);
  win_kernel<<<KE / 256, 256, 0, stream>>>(edge_index, KE, 0, KN, win1);
  e3_fill_kernel<<<KNH * KE / 256, 256, 0, stream>>>(edge_index, win1, edgesAttr, a3_gat, e3);
  {
    dim3 g(KN, KNH);
    att_kernel<<<g, 256, 0, stream>>>(s1, s2, e3, adjacency, att);
  }
  av_gemm_kernel<<<512, 256, 0, stream>>>(att, WhT, h2048);          // WMMA

  // fused edge-attr transform (elu) + a3_o reduction; We_o branch is dead
  ea3_kernel<<<64, 256, 0, stream>>>(eAh, WegT, a3_o, ea3);          // WMMA

  // edge pooling 1 + pool
  epool1_kernel<<<KN2, 256, 0, stream>>>(h2048, Wp1, bp1, hp, hph);
  gpool_kernel<<<1, 512, 0, stream>>>(hp, KN2, KD1, Wg2, bg2, hsbuf + KHID);

  // pooled-graph adjacency / edge scores
  win_kernel<<<KE / 256, 256, 0, stream>>>(edge_index, KE, 1, KN2, win2);
  pool2_scatter_kernel<<<KE / 256, 256, 0, stream>>>(edge_index, win2, ea3, adj2, e3b);

  // out_att single head
  wh2_gemm_kernel<<<16, 256, 0, stream>>>(hph, WoT, Wh2);            // WMMA
  svec2_kernel<<<1, KN2, 0, stream>>>(Wh2, a1_o, a2_o, s1o, s2o);
  att2_kernel<<<KN2, 256, 0, stream>>>(s1o, s2o, e3b, adj2, Wh2, h3);

  // edge pooling 2 + pool
  epool2_kernel<<<KN4, 128, 0, stream>>>(h3, Wp2, bp2, h4);
  gpool_kernel<<<1, 512, 0, stream>>>(h4, KN4, KOUT, Wg3, bg3, hsbuf + KHID + KD1);

  // JumpingKnowledge bi-LSTM head + fc + softmax
  head_kernel<<<1, 256, 0, stream>>>(hsbuf, Wih0, b0, Wih1, b1, fc_W, fc_b, out);
}
